// GKAN_Nodes_51187420234501
// MI455X (gfx1250) — matrix operand order repaired and checked by reading, not compiled, not run on hardware
//
#include <hip/hip_runtime.h>

// ---------------------------------------------------------------------------
// KAN-GCN for MI455X (gfx1250, wave32). Heavy math = bf16 WMMA 16x16x32 with
// fp32 accumulation; silu + spline bases computed in-register per tile (the
// [N, 8*in] expansion is never materialized). Each wave owns 32 rows x
// OTILES*16 cols; B fragments are loaded once per K-step and reused by both
// row tiles; concat-source selection is wave-uniform (no exec divergence).
// ---------------------------------------------------------------------------

typedef __bf16 bf16_t;
typedef __attribute__((ext_vector_type(16))) __bf16 v16bf;
typedef __attribute__((ext_vector_type(8)))  float  v8f;

#define D_IN  128
#define HID   64
#define N_CLS 40
#define NCOMP 8   // per input feature: 1 silu + 7 cubic B-spline bases

__device__ __forceinline__ float siluf(float x) {
  return x / (1.0f + __expf(-x));
}

// Combined 8 values for one scalar input feature.
// Uniform grid: knots t_j = -2.5 + 0.5*j, j=0..10 (grid_size=4, k=3).
// o[0] = silu(x); o[1..7] = degree-3 Cox-de Boor bases (7 of them).
__device__ __forceinline__ void kan_ext8(float x, float* o) {
  float b[10];
#pragma unroll
  for (int j = 0; j < 10; ++j) {
    float t0 = -2.5f + 0.5f * (float)j;
    float t1 = t0 + 0.5f;
    b[j] = (x >= t0 && x < t1) ? 1.0f : 0.0f;
  }
#pragma unroll
  for (int p = 1; p <= 3; ++p) {
    float inv = 2.0f / (float)p;  // 1 / (0.5 * p)
#pragma unroll
    for (int j = 0; j < 10 - p; ++j) {
      float tj   = -2.5f + 0.5f * (float)j;
      float tjp1 = -2.5f + 0.5f * (float)(j + p + 1);
      b[j] = (x - tj) * inv * b[j] + (tjp1 - x) * inv * b[j + 1];
    }
  }
  o[0] = siluf(x);
#pragma unroll
  for (int c = 0; c < 7; ++c) o[1 + c] = b[c];
}

// ---------------------------------------------------------------------------
// Pack W_ext[out, 8*in] (col order per feature: [base_w, spline_w*scaler c=0..6])
// into B-fragment layout: wfrag[(ks*OTILES+ot)*32 + lane][elem 0..15] where
// column n = lane%16, K(local) = (elem/8)*16 + (lane/16)*8 + elem%8  (ISA 7.12.2, 16-bit B).
// ---------------------------------------------------------------------------
__global__ __launch_bounds__(256)
void pack_wext_kernel(const float* __restrict__ bw, const float* __restrict__ sw,
                      const float* __restrict__ ss, bf16_t* __restrict__ wfrag,
                      int IN, int OUT, int OTILES, int total) {
  int idx = blockIdx.x * 256 + threadIdx.x;
  if (idx >= total) return;
  int elem  = idx & 15;
  int lane  = (idx >> 4) & 31;
  int rest  = idx >> 9;
  int ot    = rest % OTILES;
  int ks    = rest / OTILES;
  int n     = lane & 15;
  int khalf = lane >> 4;
  int kk = ks * 32 + (elem >> 3) * 16 + khalf * 8 + (elem & 7);   // column in W_ext
  int i  = kk >> 3;                       // input feature
  int c  = kk & 7;                        // component (0=base, 1..7=spline)
  int o  = ot * 16 + n;
  float val = 0.0f;
  if (o < OUT && i < IN) {
    if (c == 0) val = bw[(size_t)o * IN + i];
    else        val = sw[((size_t)o * IN + i) * 7 + (c - 1)] * ss[(size_t)o * IN + i];
  }
  wfrag[idx] = (bf16_t)val;
}

// ---------------------------------------------------------------------------
// Fused KANLinear: Y[N,OUT] = (silu || spline-bases)(X) @ W_ext^T via
// v_wmma_f32_16x16x32_bf16. One wave per 32-row supertile (MTILES=2 A-tiles
// sharing every B fragment); OTILES 16-col output tiles per row tile.
// A-fragment (16x32 bf16, ISA 7.12.2):
//   lane: m = lane%16, half = lane/16
//   elems 0..7  (VGPR0-3): K = half*8  + 0..7  -> feature f0+half,  comps 0..7
//   elems 8..15 (VGPR4-7): K = 16+half*8+0..7  -> feature f0+2+half, comps 0..7
// Concat segment choice is per-K-step wave-uniform (in0/in1/in2 multiples of 4).
// ---------------------------------------------------------------------------
template <int OTILES, int MTILES>
__global__ __launch_bounds__(128)
void kan_wmma_kernel(const float* __restrict__ x0, const float* __restrict__ x1,
                     const float* __restrict__ x2, int in0, int in1, int in2,
                     const bf16_t* __restrict__ wfrag, float* __restrict__ y,
                     int OUT, int n_super, int N) {
  const int wave = threadIdx.x >> 5;
  const int lane = threadIdx.x & 31;
  const int sg   = blockIdx.x * 4 + wave;       // supertile id
  if (sg >= n_super) return;
  const int row0 = sg * (16 * MTILES);
  const int m    = lane & 15;
  const int half = lane >> 4;
  const int IN   = in0 + in1 + in2;
  const int nk   = (NCOMP * IN) / 32;           // K-steps of 32
  const int ksA  = in0 >> 2;                    // segment boundaries (K-step units)
  const int ksB  = (in0 + in1) >> 2;

  int node[MTILES];
#pragma unroll
  for (int mt = 0; mt < MTILES; ++mt) {
    int r = row0 + mt * 16 + m;
    node[mt] = (r < N) ? r : (N - 1);
  }

  v8f acc[MTILES][OTILES] = {};
  const v16bf* wf = (const v16bf*)wfrag;

  for (int ks = 0; ks < nk; ++ks) {
    // ---- wave-uniform concat-segment select (SALU only, no divergence) ----
    const float* src;
    int stride, fb;
    if (ks < ksA)      { src = x0; stride = in0; fb = 0; }
    else if (ks < ksB) { src = x1; stride = in1; fb = in0; }
    else               { src = x2; stride = in2; fb = in0 + in1; }
    const int fa = ks * 4 + half - fb;          // local feature for elems 0..7
    const int fbx = fa + 2;                     // local feature for elems 8..15

    // ---- issue all loads up front: x first, then OTILES B fragments ----
    float xa[MTILES], xb[MTILES];
#pragma unroll
    for (int mt = 0; mt < MTILES; ++mt) {
      xa[mt] = src[(size_t)node[mt] * stride + fa];
      xb[mt] = src[(size_t)node[mt] * stride + fbx];
    }
    v16bf b[OTILES];
#pragma unroll
    for (int t = 0; t < OTILES; ++t)
      b[t] = wf[(size_t)(ks * OTILES + t) * 32 + lane];

    // prefetch next K-step's fragments into near cache (global_prefetch_b8)
    if (ks + 1 < nk) {
#pragma unroll
      for (int t = 0; t < OTILES; ++t)
        __builtin_prefetch((const void*)&wf[(size_t)((ks + 1) * OTILES + t) * 32 + lane], 0, 3);
    }

    // ---- long VALU chain (hides load latency; co-executes with WMMA) ----
#pragma unroll
    for (int mt = 0; mt < MTILES; ++mt) {
      float lo[8], hi[8];
      kan_ext8(xa[mt], lo);
      kan_ext8(xb[mt], hi);
      v16bf a;
#pragma unroll
      for (int c = 0; c < 8; ++c) {
        a[c]     = (bf16_t)lo[c];
        a[8 + c] = (bf16_t)hi[c];
      }
#pragma unroll
      for (int t = 0; t < OTILES; ++t) {
        acc[mt][t] = __builtin_amdgcn_wmma_f32_16x16x32_bf16(
            false, a, false, b[t], (short)0, acc[mt][t], false, false);
      }
    }
  }

  // D layout: col = lane%16 (+16*t), row = (lane/16)*8 + v
#pragma unroll
  for (int mt = 0; mt < MTILES; ++mt) {
#pragma unroll
    for (int t = 0; t < OTILES; ++t) {
      const int col = t * 16 + m;
      if (col < OUT) {
#pragma unroll
        for (int v = 0; v < 8; ++v) {
          const int r = row0 + mt * 16 + half * 8 + v;
          if (r < N) y[(size_t)r * OUT + col] = acc[mt][t][v];
        }
      }
    }
  }
}

// ------------------------------- GCN propagate -----------------------------
__global__ __launch_bounds__(256)
void deg_init_kernel(float* __restrict__ d, int N) {
  int i = blockIdx.x * 256 + threadIdx.x;
  if (i < N) d[i] = 1.0f;                 // self loop
}

__global__ __launch_bounds__(256)
void deg_accum_kernel(const int* __restrict__ col, float* __restrict__ d, int E) {
  int e = blockIdx.x * 256 + threadIdx.x;
  if (e < E) atomicAdd(&d[col[e]], 1.0f);
}

__global__ __launch_bounds__(256)
void dinv_kernel(float* __restrict__ d, int N) {
  int i = blockIdx.x * 256 + threadIdx.x;
  if (i < N) {
    float v = d[i];
    d[i] = (v > 0.0f) ? rsqrtf(fmaxf(v, 1.0f)) : 0.0f;
  }
}

// out[n][f] = tmp[n][f]*dinv[n]^2 (self loop) + conv bias  (also zero-inits out)
__global__ __launch_bounds__(256)
void prop_init_kernel(const float* __restrict__ t, const float* __restrict__ dinv,
                      const float* __restrict__ bias, float* __restrict__ out,
                      int N, int F) {
  size_t idx = (size_t)blockIdx.x * 256 + threadIdx.x;
  if (idx >= (size_t)N * F) return;
  int node = (int)(idx / F);
  int f    = (int)(idx % F);
  float di = dinv[node];
  out[idx] = t[idx] * di * di + bias[f];
}

// out[col] += tmp[row] * dinv[row]*dinv[col]; 64 feature-lanes per edge.
__global__ __launch_bounds__(256)
void prop_edges_kernel(const int* __restrict__ row, const int* __restrict__ col,
                       const float* __restrict__ t, const float* __restrict__ dinv,
                       float* __restrict__ out, int E) {
  int e = blockIdx.x * 4 + (threadIdx.x >> 6);
  int f = threadIdx.x & 63;
  if (e >= E) return;
  int r = row[e], c = col[e];
  float nrm = dinv[r] * dinv[c];
  atomicAdd(&out[(size_t)c * HID + f], t[(size_t)r * HID + f] * nrm);
}

// ------------------------------- BatchNorm ---------------------------------
__global__ void bn_zero_kernel(float* __restrict__ stats) {
  stats[threadIdx.x] = 0.0f;              // 128 threads: sum | sumsq
}

__global__ __launch_bounds__(64)
void bn_stats_kernel(const float* __restrict__ h, float* __restrict__ stats, int N) {
  int f  = threadIdx.x;                   // 0..63
  int n0 = blockIdx.x * 256;
  float s = 0.0f, s2 = 0.0f;
  for (int i = 0; i < 256; ++i) {
    int node = n0 + i;
    if (node < N) {
      float v = h[(size_t)node * HID + f];
      s += v; s2 += v * v;
    }
  }
  atomicAdd(&stats[f], s);
  atomicAdd(&stats[HID + f], s2);
}

__global__ __launch_bounds__(256)
void bn_apply_kernel(float* __restrict__ h, const float* __restrict__ stats,
                     const float* __restrict__ gamma, const float* __restrict__ beta,
                     int N) {
  size_t idx = (size_t)blockIdx.x * 256 + threadIdx.x;
  if (idx >= (size_t)N * HID) return;
  int f = (int)(idx % HID);
  float inv_n = 1.0f / (float)N;
  float mu  = stats[f] * inv_n;
  float var = stats[HID + f] * inv_n - mu * mu;
  h[idx] = (h[idx] - mu) * rsqrtf(var + 1e-5f) * gamma[f] + beta[f];
}

// ---------------------------------------------------------------------------
extern "C" void kernel_launch(void* const* d_in, const int* in_sizes, int n_in,
                              void* d_out, int out_size, void* d_ws, size_t ws_size,
                              hipStream_t stream) {
  (void)n_in; (void)out_size; (void)ws_size;
  const float* x      = (const float*)d_in[0];
  const float* bw0    = (const float*)d_in[1];
  const float* sw0    = (const float*)d_in[2];
  const float* ss0    = (const float*)d_in[3];
  const float* cbias0 = (const float*)d_in[4];
  const float* gamma0 = (const float*)d_in[5];
  const float* beta0  = (const float*)d_in[6];
  const float* bw1    = (const float*)d_in[7];
  const float* sw1    = (const float*)d_in[8];
  const float* ss1    = (const float*)d_in[9];
  const float* cbias1 = (const float*)d_in[10];
  const float* gamma1 = (const float*)d_in[11];
  const float* beta1  = (const float*)d_in[12];
  const float* bwo    = (const float*)d_in[13];
  const float* swo    = (const float*)d_in[14];
  const float* sso    = (const float*)d_in[15];
  const int*   ei     = (const int*)d_in[16];

  const int N = in_sizes[0] / D_IN;       // 100000
  const int E = in_sizes[16] / 2;         // 1600000
  const int* erow = ei;
  const int* ecol = ei + E;

  // ---- workspace carve (256B aligned) ----
  char* ws = (char*)d_ws;
  size_t off = 0;
  auto carve = [&](size_t bytes) -> char* {
    char* p = ws + off;
    off = (off + bytes + 255) & ~(size_t)255;
    return p;
  };
  float* tmp   = (float*)carve((size_t)N * HID * 4);  // pre-propagate activations
  float* h0    = (float*)carve((size_t)N * HID * 4);  // skip 1
  float* h1    = (float*)carve((size_t)N * HID * 4);  // skip 2
  float* dinv  = (float*)carve((size_t)N * 4);
  float* stats = (float*)carve(2 * HID * 4);

  const int nk0   = (NCOMP * D_IN) / 32;              // 32 K-steps
  const int nk1   = (NCOMP * HID) / 32;               // 16
  const int d_cat = D_IN + 2 * HID;                   // 256
  const int nko   = (NCOMP * d_cat) / 32;             // 64
  bf16_t* wf0 = (bf16_t*)carve((size_t)nk0 * 4 * 512 * sizeof(bf16_t));
  bf16_t* wf1 = (bf16_t*)carve((size_t)nk1 * 4 * 512 * sizeof(bf16_t));
  bf16_t* wfo = (bf16_t*)carve((size_t)nko * 3 * 512 * sizeof(bf16_t));

  // ---- pack combined (base||spline) weights into B-fragment layout ----
  {
    int t0 = nk0 * 4 * 512;
    pack_wext_kernel<<<(t0 + 255) / 256, 256, 0, stream>>>(bw0, sw0, ss0, wf0, D_IN, HID, 4, t0);
    int t1 = nk1 * 4 * 512;
    pack_wext_kernel<<<(t1 + 255) / 256, 256, 0, stream>>>(bw1, sw1, ss1, wf1, HID, HID, 4, t1);
    int to = nko * 3 * 512;
    pack_wext_kernel<<<(to + 255) / 256, 256, 0, stream>>>(bwo, swo, sso, wfo, d_cat, N_CLS, 3, to);
  }

  // ---- shared degree / normalization ----
  deg_init_kernel<<<(N + 255) / 256, 256, 0, stream>>>(dinv, N);
  deg_accum_kernel<<<(E + 255) / 256, 256, 0, stream>>>(ecol, dinv, E);
  dinv_kernel<<<(N + 255) / 256, 256, 0, stream>>>(dinv, N);

  const int n_rt      = (N + 15) / 16;                // 6250 row tiles
  const int n_super   = (n_rt + 1) / 2;               // 3125 (MTILES=2)
  const int kan_blocks = (n_super + 3) / 4;           // 4 waves / block
  const int nf_blocks  = (int)(((size_t)N * HID + 255) / 256);

  // ---- layer 0: KANLinear -> GCN propagate (+bias) -> BN ----
  kan_wmma_kernel<4, 2><<<kan_blocks, 128, 0, stream>>>(x, x, x, D_IN, 0, 0,
                                                        wf0, tmp, HID, n_super, N);
  prop_init_kernel<<<nf_blocks, 256, 0, stream>>>(tmp, dinv, cbias0, h0, N, HID);
  prop_edges_kernel<<<(E + 3) / 4, 256, 0, stream>>>(erow, ecol, tmp, dinv, h0, E);
  bn_zero_kernel<<<1, 2 * HID, 0, stream>>>(stats);
  bn_stats_kernel<<<(N + 255) / 256, HID, 0, stream>>>(h0, stats, N);
  bn_apply_kernel<<<nf_blocks, 256, 0, stream>>>(h0, stats, gamma0, beta0, N);

  // ---- layer 1 ----
  kan_wmma_kernel<4, 2><<<kan_blocks, 128, 0, stream>>>(h0, h0, h0, HID, 0, 0,
                                                        wf1, tmp, HID, n_super, N);
  prop_init_kernel<<<nf_blocks, 256, 0, stream>>>(tmp, dinv, cbias1, h1, N, HID);
  prop_edges_kernel<<<(E + 3) / 4, 256, 0, stream>>>(erow, ecol, tmp, dinv, h1, E);
  bn_zero_kernel<<<1, 2 * HID, 0, stream>>>(stats);
  bn_stats_kernel<<<(N + 255) / 256, HID, 0, stream>>>(h1, stats, N);
  bn_apply_kernel<<<nf_blocks, 256, 0, stream>>>(h1, stats, gamma1, beta1, N);

  // ---- output head on virtual concat [x || h0 || h1] -> d_out [N, 40] ----
  kan_wmma_kernel<3, 2><<<kan_blocks, 128, 0, stream>>>(x, h0, h1, D_IN, HID, HID,
                                                        wfo, (float*)d_out, N_CLS, n_super, N);
}
